// MAB_43181601194591
// MI455X (gfx1250) — compile-verified, matrix operand order checked by reading, not run on hardware
//
#include <hip/hip_runtime.h>

// ---------------------------------------------------------------------------
// MAB (Set-Transformer Multihead Attention Block) for MI455X / gfx1250.
// All GEMMs on v_wmma_f32_16x16x32_f16; weights pre-converted to f16 once;
// K/V/H intermediates stored f16 (V transposed) so WMMA operands stream as
// contiguous b128 loads.  Softmax / LayerNorm / SiLU / residuals in f32.
// ---------------------------------------------------------------------------

typedef __attribute__((ext_vector_type(16))) _Float16 v16h;
typedef __attribute__((ext_vector_type(8)))  float    v8f;

#define DIMV 1024
#define BATCH 4
#define NQ 1024
#define NK 1024
#define HEADS 16
#define DH 64
#define ROWS (BATCH * NQ)   // 4096

__device__ __forceinline__ v8f wmma_f16(v16h a, v16h b, v8f c) {
  return __builtin_amdgcn_wmma_f32_16x16x32_f16(
      false, a, false, b, (short)0, c, false, false);
}

// Fragment loaders: lane's 16 K-elements; p already includes the lane's row
// and half-wave K offset (lhalf*8); elems 8..15 come from +16 elements.
__device__ __forceinline__ v16h frag_contig(const float* p) {
  v16h f;
#pragma unroll
  for (int e = 0; e < 8; ++e) f[e] = (_Float16)p[e];
#pragma unroll
  for (int e = 0; e < 8; ++e) f[8 + e] = (_Float16)p[16 + e];
  return f;
}
__device__ __forceinline__ v16h frag_contig(const _Float16* p) {
  v16h f;
#pragma unroll
  for (int e = 0; e < 8; ++e) f[e] = p[e];         // 16B -> b128
#pragma unroll
  for (int e = 0; e < 8; ++e) f[8 + e] = p[16 + e]; // 16B -> b128
  return f;
}

// ---------------------------------------------------------------------------
// Y = X[4096,Kd] * Wh[C,Kd]^T + bias (+res).  Wh is f16.
// Wave tile 32(M) x 64(N): 8 WMMA per K-step, B frags reused across 2 A rows.
// MODE 0: f32 out (+res);  MODE 1: f16 out;  MODE 2: f16 out transposed
// per batch:  Y[b][c][row%1024]  (for V^T).
// ---------------------------------------------------------------------------
template <typename TA, int MODE>
__global__ __launch_bounds__(256) void wmma_gemm_kernel(
    const TA* __restrict__ X, const _Float16* __restrict__ Wh,
    const float* __restrict__ bias, const float* __restrict__ res,
    void* __restrict__ Yv, int Kd, int C) {
  const int lane  = threadIdx.x & 31;
  const int wave  = threadIdx.x >> 5;
  const int lhalf = lane >> 4;
  const int lidx  = lane & 15;
  const int row0  = blockIdx.y * 64 + (wave >> 2) * 32;
  const int col0  = blockIdx.x * 256 + (wave & 3) * 64;

  v8f acc[2][4] = {};
  const TA* ap0 = X + (size_t)(row0 + lidx) * Kd + lhalf * 8;
  const TA* ap1 = X + (size_t)(row0 + 16 + lidx) * Kd + lhalf * 8;
  const _Float16* bp[4];
#pragma unroll
  for (int t = 0; t < 4; ++t)
    bp[t] = Wh + (size_t)(col0 + t * 16 + lidx) * Kd + lhalf * 8;

  for (int k0 = 0; k0 < Kd; k0 += 32) {
    __builtin_prefetch(ap0 + k0 + 256, 0, 1);   // global_prefetch_b8
    __builtin_prefetch(ap1 + k0 + 256, 0, 1);
    v16h a0 = frag_contig(ap0 + k0);
    v16h a1 = frag_contig(ap1 + k0);
#pragma unroll
    for (int t = 0; t < 4; ++t) {
      v16h b = frag_contig(bp[t] + k0);
      acc[0][t] = wmma_f16(a0, b, acc[0][t]);
      acc[1][t] = wmma_f16(a1, b, acc[1][t]);
    }
  }

#pragma unroll
  for (int i = 0; i < 2; ++i) {
#pragma unroll
    for (int t = 0; t < 4; ++t) {
      const int c = col0 + t * 16 + lidx;
      const float bv = bias[c];
#pragma unroll
      for (int r = 0; r < 8; ++r) {
        const int row = row0 + i * 16 + r + lhalf * 8;  // C/D: lanes>=16 -> M+8
        float v = acc[i][t][r] + bv;
        if (MODE == 0) {
          float* Y = (float*)Yv;
          const size_t off = (size_t)row * C + c;
          if (res) v += res[off];
          Y[off] = v;
        } else if (MODE == 1) {
          ((_Float16*)Yv)[(size_t)row * C + c] = (_Float16)v;
        } else {
          const int bb = row >> 10, n = row & 1023;     // per-batch transpose
          ((_Float16*)Yv)[((size_t)bb * DIMV + c) * NK + n] = (_Float16)v;
        }
      }
    }
  }
}

// ---------------------------------------------------------------------------
// SwiGLU:  H = silu(X*W12[0:1024]^T + b1) * (X*W12[1024:]^T + b2),  f16 in/out.
// Wave tile 32(M) x 16(N) with paired accumulators.
// ---------------------------------------------------------------------------
__global__ __launch_bounds__(256) void swiglu_gemm_kernel(
    const _Float16* __restrict__ Xh, const _Float16* __restrict__ W12h,
    const float* __restrict__ b12, _Float16* __restrict__ Hout) {
  const int lane  = threadIdx.x & 31;
  const int wave  = threadIdx.x >> 5;
  const int lhalf = lane >> 4;
  const int lidx  = lane & 15;
  const int col0  = blockIdx.x * 128 + wave * 16;
  const int row0  = blockIdx.y * 32;

  v8f acc1[2] = {}, acc2[2] = {};
  const _Float16* ap0 = Xh + (size_t)(row0 + lidx) * DIMV + lhalf * 8;
  const _Float16* ap1 = Xh + (size_t)(row0 + 16 + lidx) * DIMV + lhalf * 8;
  const _Float16* bp1 = W12h + (size_t)(col0 + lidx) * DIMV + lhalf * 8;
  const _Float16* bp2 = W12h + (size_t)(DIMV + col0 + lidx) * DIMV + lhalf * 8;

  for (int k0 = 0; k0 < DIMV; k0 += 32) {
    v16h a0 = frag_contig(ap0 + k0);
    v16h a1 = frag_contig(ap1 + k0);
    v16h b1 = frag_contig(bp1 + k0);
    acc1[0] = wmma_f16(a0, b1, acc1[0]);
    acc1[1] = wmma_f16(a1, b1, acc1[1]);
    v16h b2 = frag_contig(bp2 + k0);
    acc2[0] = wmma_f16(a0, b2, acc2[0]);
    acc2[1] = wmma_f16(a1, b2, acc2[1]);
  }

  const int c = col0 + lidx;
  const float b1 = b12[c], b2 = b12[DIMV + c];
#pragma unroll
  for (int i = 0; i < 2; ++i) {
#pragma unroll
    for (int r = 0; r < 8; ++r) {
      const int row = row0 + i * 16 + r + lhalf * 8;
      const float x1 = acc1[i][r] + b1;
      const float x2 = acc2[i][r] + b2;
      const float sil = x1 / (1.0f + __expf(-x1));
      Hout[(size_t)row * DIMV + c] = (_Float16)(sil * x2);
    }
  }
}

// ---------------------------------------------------------------------------
// Attention per (q-tile of 16, batch, head):
//   S = Qh*Kh^T / 32 -> LDS; softmax (writes A); O = Qh + P*Vh.
// Kh: f16 [B][key][1024].  Vt: f16 [B][dim][key] (transposed -> contiguous B).
// ---------------------------------------------------------------------------
__global__ __launch_bounds__(256) void attn_kernel(
    const float* __restrict__ Qp, const _Float16* __restrict__ Kh,
    const _Float16* __restrict__ Vt, float* __restrict__ Aout,
    float* __restrict__ Opre) {
  __shared__ float S[16 * NK];                     // 64 KB
  const int qt = blockIdx.x, b = blockIdx.y, h = blockIdx.z;
  const int lane  = threadIdx.x & 31;
  const int wave  = threadIdx.x >> 5;
  const int lhalf = lane >> 4;
  const int lidx  = lane & 15;
  const int q0 = qt * 16;
  const size_t baseB = (size_t)b * NK * DIMV;
  const int hoff = h * DH;

  // Phase 1: scores.  Q frags hoisted (K = DH = 64 -> 2 frags).
  const float* aq = Qp + baseB + (size_t)(q0 + lidx) * DIMV + hoff + lhalf * 8;
  const v16h a0 = frag_contig(aq);
  const v16h a1 = frag_contig(aq + 32);
  const float scale = 1.0f / 32.0f;                // 1/sqrt(DIM_V)
  for (int t = wave; t < NK / 16; t += 8) {
    const int n0 = t * 16;
    const _Float16* bk = Kh + baseB + (size_t)(n0 + lidx) * DIMV + hoff + lhalf * 8;
    v8f acc = {};
    acc = wmma_f16(a0, frag_contig(bk), acc);
    acc = wmma_f16(a1, frag_contig(bk + 32), acc);
#pragma unroll
    for (int r = 0; r < 8; ++r)
      S[(r + lhalf * 8) * NK + n0 + lidx] = acc[r] * scale;
  }
  __syncthreads();

  // Phase 2: softmax.  Wave w owns rows 2w, 2w+1; lane-parallel over NK.
#pragma unroll
  for (int rr = 0; rr < 2; ++rr) {
    const int m = wave * 2 + rr;
    float* srow = &S[m * NK];
    float mx = -3.4e38f;
    for (int c = lane; c < NK; c += 32) mx = fmaxf(mx, srow[c]);
#pragma unroll
    for (int o = 16; o > 0; o >>= 1) mx = fmaxf(mx, __shfl_xor(mx, o, 32));
    float sum = 0.0f;
    for (int c = lane; c < NK; c += 32) {
      const float e = __expf(srow[c] - mx);
      srow[c] = e;
      sum += e;
    }
#pragma unroll
    for (int o = 16; o > 0; o >>= 1) sum += __shfl_xor(sum, o, 32);
    const float inv = 1.0f / sum;
    float* arow = Aout + ((size_t)(h * BATCH + b) * NQ + (q0 + m)) * NK;
    for (int c = lane; c < NK; c += 32) {
      const float p = srow[c] * inv;
      srow[c] = p;
      arow[c] = p;                                 // coalesced 256 MB stream
    }
  }
  __syncthreads();

  // Phase 3: O = Q + P*V.  Waves 0..3 each own a 16-col tile of dh=64.
  if (wave < 4) {
    const int n0 = wave * 16;
    v8f acc = {};
    const float* sA = &S[lidx * NK + lhalf * 8];   // LDS, contiguous K
    const _Float16* vB =
        Vt + ((size_t)b * DIMV + hoff + n0 + lidx) * NK + lhalf * 8;
    for (int k0 = 0; k0 < NK; k0 += 32) {
      v16h a = frag_contig(sA + k0);               // ds_load + cvt
      v16h bf = frag_contig(vB + k0);              // contiguous f16 b128
      acc = wmma_f16(a, bf, acc);
    }
#pragma unroll
    for (int r = 0; r < 8; ++r) {
      const int m = r + lhalf * 8;
      const size_t off = baseB + (size_t)(q0 + m) * DIMV + hoff + n0 + lidx;
      Opre[off] = acc[r] + Qp[off];                // residual pre-concat
    }
  }
}

// ---------------------------------------------------------------------------
// LayerNorm over last dim; optional extra f16 copy of the output.
// ---------------------------------------------------------------------------
__global__ __launch_bounds__(256) void ln_kernel(
    const float* __restrict__ X, const float* __restrict__ g,
    const float* __restrict__ beta, float* __restrict__ Y,
    _Float16* __restrict__ Y16, int D) {
  __shared__ float red[16];
  const int lane = threadIdx.x & 31, wave = threadIdx.x >> 5;
  const float* x = X + (size_t)blockIdx.x * D;

  float s = 0.0f;
  for (int c = threadIdx.x; c < D; c += 256) s += x[c];
#pragma unroll
  for (int o = 16; o > 0; o >>= 1) s += __shfl_xor(s, o, 32);
  if (lane == 0) red[wave] = s;
  __syncthreads();
  if (threadIdx.x == 0) {
    float t = 0.0f;
    for (int i = 0; i < 8; ++i) t += red[i];
    red[8] = t;
  }
  __syncthreads();
  const float mu = red[8] / D;

  float v = 0.0f;
  for (int c = threadIdx.x; c < D; c += 256) {
    const float d = x[c] - mu;
    v += d * d;
  }
#pragma unroll
  for (int o = 16; o > 0; o >>= 1) v += __shfl_xor(v, o, 32);
  if (lane == 0) red[wave] = v;
  __syncthreads();
  if (threadIdx.x == 0) {
    float t = 0.0f;
    for (int i = 0; i < 8; ++i) t += red[i];
    red[9] = t;
  }
  __syncthreads();
  const float rstd = rsqrtf(red[9] / D + 1e-5f);
  for (int c = threadIdx.x; c < D; c += 256) {
    const float val = (x[c] - mu) * rstd * g[c] + beta[c];
    Y[(size_t)blockIdx.x * D + c] = val;
    if (Y16) Y16[(size_t)blockIdx.x * D + c] = (_Float16)val;
  }
}

// ---------------------------------------------------------------------------
// One-shot f32 -> f16 weight conversion (amortized over 60 GFLOP of GEMM).
// ---------------------------------------------------------------------------
__global__ __launch_bounds__(256) void cvt_f16_kernel(
    const float* __restrict__ src, _Float16* __restrict__ dst, int n) {
  const int i = (blockIdx.x * 256 + threadIdx.x) * 4;
  if (i < n) {
#pragma unroll
    for (int j = 0; j < 4; ++j) dst[i + j] = (_Float16)src[i + j];
  }
}

// ---------------------------------------------------------------------------
extern "C" void kernel_launch(void* const* d_in, const int* in_sizes, int n_in,
                              void* d_out, int out_size, void* d_ws, size_t ws_size,
                              hipStream_t stream) {
  const float* Q    = (const float*)d_in[0];
  const float* K    = (const float*)d_in[1];
  const float* Wq   = (const float*)d_in[2];
  const float* bq   = (const float*)d_in[3];
  const float* Wk   = (const float*)d_in[4];
  const float* bk   = (const float*)d_in[5];
  const float* Wv   = (const float*)d_in[6];
  const float* bv   = (const float*)d_in[7];
  const float* ln0g = (const float*)d_in[8];
  const float* ln0b = (const float*)d_in[9];
  const float* ln1g = (const float*)d_in[10];
  const float* ln1b = (const float*)d_in[11];
  const float* W12  = (const float*)d_in[12];
  const float* b12  = (const float*)d_in[13];
  const float* W3   = (const float*)d_in[14];
  const float* b3   = (const float*)d_in[15];

  float* Oout = (float*)d_out;                     // [4,1024,1024]
  float* Aout = Oout + (size_t)BATCH * NQ * DIMV;  // [64,1024,1024]

  const size_t MF = 1024 * 1024;                   // 1M
  float* wsf = (float*)d_ws;
  float*    Qp     = wsf;                              // 4M f32
  _Float16* Kh     = (_Float16*)(wsf + 4 * MF);        // 4M f16
  _Float16* Vt     = (_Float16*)(wsf + 6 * MF);        // 4M f16 (transposed)
  float*    Opre   = wsf + 8 * MF;                     // 4M f32
  float*    Oln    = wsf + 12 * MF;                    // 4M f32
  _Float16* Wq16   = (_Float16*)(wsf + 16 * MF);       // 1M f16
  _Float16* Wk16   = Wq16 + 1 * MF;
  _Float16* Wv16   = Wq16 + 2 * MF;
  _Float16* W12_16 = Wq16 + 3 * MF;                    // 2M f16
  _Float16* W3_16  = Wq16 + 5 * MF;                    // 1M f16
  _Float16* Hbuf   = Kh;                               // Kh dead after attn
  _Float16* Oln16  = Vt;                               // Vt dead after attn
  float*    Ybuf   = Qp;                               // Qp dead after attn

  dim3 blk(256);
  dim3 cvt_g(MF / 1024);
  cvt_f16_kernel<<<cvt_g, blk, 0, stream>>>(Wq, Wq16, (int)MF);
  cvt_f16_kernel<<<cvt_g, blk, 0, stream>>>(Wk, Wk16, (int)MF);
  cvt_f16_kernel<<<cvt_g, blk, 0, stream>>>(Wv, Wv16, (int)MF);
  cvt_f16_kernel<<<dim3(2 * MF / 1024), blk, 0, stream>>>(W12, W12_16, (int)(2 * MF));
  cvt_f16_kernel<<<cvt_g, blk, 0, stream>>>(W3, W3_16, (int)MF);

  dim3 ggrid(DIMV / 256, ROWS / 64);               // (4, 64)
  // Projections: Q -> f32 (residual needs it); K -> f16; V -> f16 transposed.
  wmma_gemm_kernel<float, 0><<<ggrid, blk, 0, stream>>>(Q, Wq16, bq, nullptr, Qp, DIMV, DIMV);
  wmma_gemm_kernel<float, 1><<<ggrid, blk, 0, stream>>>(K, Wk16, bk, nullptr, Kh, DIMV, DIMV);
  wmma_gemm_kernel<float, 2><<<ggrid, blk, 0, stream>>>(K, Wv16, bv, nullptr, Vt, DIMV, DIMV);

  attn_kernel<<<dim3(NQ / 16, BATCH, HEADS), blk, 0, stream>>>(Qp, Kh, Vt, Aout, Opre);

  ln_kernel<<<dim3(ROWS), blk, 0, stream>>>(Opre, ln0g, ln0b, Oln, Oln16, DIMV);

  swiglu_gemm_kernel<<<dim3(DIMV / 128, ROWS / 32), blk, 0, stream>>>(Oln16, W12_16, b12, Hbuf);

  wmma_gemm_kernel<_Float16, 0><<<ggrid, blk, 0, stream>>>(Hbuf, W3_16, b3, Oln, Ybuf, DIMV, DIMV);

  ln_kernel<<<dim3(ROWS), blk, 0, stream>>>(Ybuf, ln1g, ln1b, Oout, nullptr, DIMV);
}
